// GNN_Model_13735305413435
// MI455X (gfx1250) — compile-verified, matrix operand order checked by reading, not run on hardware
//
#include <hip/hip_runtime.h>
#include <hip/hip_bf16.h>

// GIN conv + 3x(Dense+BN+ReLU) + maxpool + head for MI455X (gfx1250).
//  - fp16 WMMA (v_wmma_f32_16x16x32_f16), f32 accumulate.
//  - `a` (134MB f32) converted to f16 during LDS staging with vector b128
//    loads + packed b128 ds stores (binary adjacency -> exact in f16).
//  - x and weights pre-transposed to f16 [col][k] so GEMM B tiles (and MLP
//    A tiles) stage via GLOBAL_LOAD_ASYNC_TO_LDS_B128 (ASYNCcnt path).
//  - BN+bias folded to per-feature scale/bias, ReLU fused in epilogue.

typedef __attribute__((ext_vector_type(16))) _Float16 v16h;
typedef __attribute__((ext_vector_type(8)))  _Float16 v8h;
typedef __attribute__((ext_vector_type(8)))  float    v8f;
typedef int v4i_vec __attribute__((vector_size(16)));   // pointee type the async builtin expects

#define BN_EPS 1e-3f

// ---- CDNA5 async-to-LDS availability probes (compile-time) ----
#if __has_builtin(__builtin_amdgcn_global_load_async_to_lds_b128)
  #define ASYNC_LDS 1
#else
  #define ASYNC_LDS 0
  #if defined(__HIP_DEVICE_COMPILE__)
    #warning "gfx1250: __builtin_amdgcn_global_load_async_to_lds_b128 NOT available; falling back to sync staging"
  #endif
#endif
#if __has_builtin(__builtin_amdgcn_s_wait_asynccnt)
  #define HAVE_WAIT_ASYNC_BUILTIN 1
#else
  #define HAVE_WAIT_ASYNC_BUILTIN 0
#endif

// Copy 16 bytes global -> LDS via the async DMA path when available.
__device__ __forceinline__ void async_b128(const void* g, void* l) {
#if ASYNC_LDS
    typedef __attribute__((address_space(1))) v4i_vec* gp_t;
    typedef __attribute__((address_space(3))) v4i_vec* lp_t;
    __builtin_amdgcn_global_load_async_to_lds_b128(
        (gp_t)const_cast<void*>(g), (lp_t)l, 0, 0);
#else
    *(v8h*)l = *(const v8h*)g;
#endif
}

__device__ __forceinline__ void wait_async_lds() {
#if ASYNC_LDS
  #if HAVE_WAIT_ASYNC_BUILTIN
    __builtin_amdgcn_s_wait_asynccnt(0);
  #else
    asm volatile("s_wait_asynccnt 0x0" ::: "memory");
  #endif
#endif
}

// WMMA f16 fragment load from an LDS tile stored row-major [idx][k] (+pad),
// where idx = M for A fragments / N for B fragments. CDNA5 16-bit layout:
//   element e of lane L: k = (e/8)*16 + (L/16)*8 + (e%8)
// -> two aligned 16B ds loads per lane.
__device__ __forceinline__ v16h frag_load(const _Float16* rowbase, int laneHalf) {
    const v8h lo = *(const v8h*)(rowbase + laneHalf * 8);
    const v8h hi = *(const v8h*)(rowbase + 16 + laneHalf * 8);
    return __builtin_shufflevector(lo, hi, 0,1,2,3,4,5,6,7,8,9,10,11,12,13,14,15);
}

// ---------------- Setup: f32 [R][C] -> f16 transposed [C][R] ----------------
// (batched via blockIdx.z with element strides; strides 0 + grid.z=1 for weights)
__global__ __launch_bounds__(256) void cvt_transpose_kernel(
    const float* __restrict__ src, _Float16* __restrict__ dst,
    int R, int C, size_t srcBatchStride, size_t dstBatchStride)
{
    __shared__ float tile[32][33];
    const float* s = src + (size_t)blockIdx.z * srcBatchStride;
    _Float16*    d = dst + (size_t)blockIdx.z * dstBatchStride;
    const int r0 = blockIdx.x * 32, c0 = blockIdx.y * 32;
    const int tr = threadIdx.x >> 5;      // 0..7
    const int tc = threadIdx.x & 31;
    #pragma unroll
    for (int i = 0; i < 4; ++i) {
        int r = tr + i * 8;
        tile[r][tc] = s[(size_t)(r0 + r) * C + c0 + tc];
    }
    __syncthreads();
    #pragma unroll
    for (int i = 0; i < 4; ++i) {
        int c = tr + i * 8;
        d[(size_t)(c0 + c) * R + r0 + tc] = (_Float16)tile[tc][c];
    }
}

// -------------------- Kernel 1: h0 = a @ x + (1+eps)*x --------------------
// Per batch [1024x1024]*[1024x128]. Block tile 64(M)x64(F), 8 waves (4x2),
// each wave: 1 A frag, 2 B frags, 2 WMMAs per 32-wide K step.
__global__ __launch_bounds__(256) void gin_agg_kernel(
    const float* __restrict__ a, const _Float16* __restrict__ xT,
    const float* __restrict__ x, const float* __restrict__ epsp,
    _Float16* __restrict__ h0)
{
    __shared__ __align__(16) _Float16 As[64][40];   // [m][k] f16 (cvt from f32 a)
    __shared__ __align__(16) _Float16 Bs[64][40];   // [f][k] f16 (async from xT)

    const int tid  = threadIdx.x;
    const int lane = tid & 31;
    const int wave = tid >> 5;
    const int wm   = wave & 3;            // M sub-tile (x16)
    const int wn   = wave >> 2;           // N sub-tile (x32)
    const int b    = blockIdx.z;
    const int i0   = blockIdx.x * 64;
    const int f0   = blockIdx.y * 64;

    // staging chunk: 8 contiguous k-elements per thread
    const int srow = tid >> 2;            // 0..63
    const int scol = (tid & 3) * 8;       // 0,8,16,24

    const float*    Ag  = a  + (size_t)b * 1024 * 1024 + (size_t)(i0 + srow) * 1024 + scol;
    const _Float16* Bg  = xT + (size_t)b * 128 * 1024 + (size_t)(f0 + srow) * 1024 + scol;

    v8f acc0 = {}, acc1 = {};
    for (int k0 = 0; k0 < 1024; k0 += 32) {
        if (k0 + 32 < 1024)
            __builtin_prefetch(Ag + k0 + 32, 0, 3);       // global_prefetch_b8
        __syncthreads();
        // B tile: pure f16 byte copy -> async DMA to LDS
        async_b128(Bg + k0, &Bs[srow][scol]);
        // A tile: two b128 f32 loads -> cvt -> one packed b128 ds store
        {
            const float4* gp = (const float4*)(Ag + k0);
            float4 fa = gp[0], fb = gp[1];
            v8h hv;
            hv[0] = (_Float16)fa.x; hv[1] = (_Float16)fa.y;
            hv[2] = (_Float16)fa.z; hv[3] = (_Float16)fa.w;
            hv[4] = (_Float16)fb.x; hv[5] = (_Float16)fb.y;
            hv[6] = (_Float16)fb.z; hv[7] = (_Float16)fb.w;
            *(v8h*)&As[srow][scol] = hv;
        }
        wait_async_lds();
        __syncthreads();

        v16h afrag = frag_load(&As[wm * 16 + (lane & 15)][0], lane >> 4);
        v16h bf0   = frag_load(&Bs[wn * 32 + (lane & 15)][0], lane >> 4);
        v16h bf1   = frag_load(&Bs[wn * 32 + 16 + (lane & 15)][0], lane >> 4);
        acc0 = __builtin_amdgcn_wmma_f32_16x16x32_f16(false, afrag, false, bf0,
                                                      (short)0, acc0, false, false);
        acc1 = __builtin_amdgcn_wmma_f32_16x16x32_f16(false, afrag, false, bf1,
                                                      (short)0, acc1, false, false);
    }

    const float ope = 1.0f + epsp[0];
    #pragma unroll
    for (int r = 0; r < 8; ++r) {
        int row  = i0 + wm * 16 + (lane >> 4) * 8 + r;
        int col0 = f0 + wn * 32 + (lane & 15);
        size_t xi0 = ((size_t)b * 1024 + row) * 128 + col0;
        size_t xi1 = xi0 + 16;
        h0[xi0] = (_Float16)(acc0[r] + ope * x[xi0]);
        h0[xi1] = (_Float16)(acc1[r] + ope * x[xi1]);
    }
}

// ---------- Kernel 2: out = relu((Ain @ W)*s + t), all-async staging ----------
// Ain f16 [32768 x K] row-major; Wt f16 [256 x K] (pre-transposed).
__global__ __launch_bounds__(256) void mlp_layer_kernel(
    const _Float16* __restrict__ Ain, int K,
    const _Float16* __restrict__ Wt,
    const float* __restrict__ bias,
    const float* __restrict__ gam, const float* __restrict__ bet,
    const float* __restrict__ mu,  const float* __restrict__ var,
    _Float16* __restrict__ out)
{
    __shared__ __align__(16) _Float16 As[64][40];   // [row][k]
    __shared__ __align__(16) _Float16 Bs[64][40];   // [o][k]

    const int tid  = threadIdx.x;
    const int lane = tid & 31;
    const int wave = tid >> 5;
    const int wm   = wave & 3;
    const int wn   = wave >> 2;
    const int r0   = blockIdx.x * 64;
    const int o0   = blockIdx.y * 64;
    const int O    = 256;

    const int srow = tid >> 2;
    const int scol = (tid & 3) * 8;

    const _Float16* Agl = Ain + (size_t)(r0 + srow) * K + scol;
    const _Float16* Bgl = Wt  + (size_t)(o0 + srow) * K + scol;

    v8f acc0 = {}, acc1 = {};
    for (int k0 = 0; k0 < K; k0 += 32) {
        __syncthreads();
        async_b128(Agl + k0, &As[srow][scol]);
        async_b128(Bgl + k0, &Bs[srow][scol]);
        wait_async_lds();
        __syncthreads();

        v16h afrag = frag_load(&As[wm * 16 + (lane & 15)][0], lane >> 4);
        v16h bf0   = frag_load(&Bs[wn * 32 + (lane & 15)][0], lane >> 4);
        v16h bf1   = frag_load(&Bs[wn * 32 + 16 + (lane & 15)][0], lane >> 4);
        acc0 = __builtin_amdgcn_wmma_f32_16x16x32_f16(false, afrag, false, bf0,
                                                      (short)0, acc0, false, false);
        acc1 = __builtin_amdgcn_wmma_f32_16x16x32_f16(false, afrag, false, bf1,
                                                      (short)0, acc1, false, false);
    }

    const int oc0 = o0 + wn * 32 + (lane & 15);
    const int oc1 = oc0 + 16;
    const float s0 = gam[oc0] * rsqrtf(var[oc0] + BN_EPS);
    const float t0 = (bias[oc0] - mu[oc0]) * s0 + bet[oc0];
    const float s1 = gam[oc1] * rsqrtf(var[oc1] + BN_EPS);
    const float t1 = (bias[oc1] - mu[oc1]) * s1 + bet[oc1];
    #pragma unroll
    for (int r = 0; r < 8; ++r) {
        int row = r0 + wm * 16 + (lane >> 4) * 8 + r;
        out[(size_t)row * O + oc0] = (_Float16)fmaxf(acc0[r] * s0 + t0, 0.0f);
        out[(size_t)row * O + oc1] = (_Float16)fmaxf(acc1[r] * s1 + t1, 0.0f);
    }
}

// ---------------- Kernel 3: global max pool over N + head Dense(3) ----------------
__global__ __launch_bounds__(256) void pool_head_kernel(
    const _Float16* __restrict__ h3, const float* __restrict__ wd,
    const float* __restrict__ bd, float* __restrict__ out)
{
    const int b = blockIdx.x;
    const int o = threadIdx.x;
    __shared__ float pooled[256];

    const _Float16* p = h3 + ((size_t)b * 1024) * 256 + o;
    float mx = -3.0e38f;
    for (int n = 0; n < 1024; ++n)
        mx = fmaxf(mx, (float)p[(size_t)n * 256]);
    pooled[o] = mx;
    __syncthreads();

    if (o < 3) {
        float s = bd[o];
        for (int k = 0; k < 256; ++k)
            s += pooled[k] * wd[k * 3 + o];
        out[b * 3 + o] = s;
    }
}

extern "C" void kernel_launch(void* const* d_in, const int* in_sizes, int n_in,
                              void* d_out, int out_size, void* d_ws, size_t ws_size,
                              hipStream_t stream)
{
    (void)in_sizes; (void)n_in; (void)out_size; (void)ws_size;
    const float* x   = (const float*)d_in[0];
    const float* a   = (const float*)d_in[1];
    const float* eps = (const float*)d_in[2];
    const float* w1  = (const float*)d_in[3];
    const float* b1  = (const float*)d_in[4];
    const float* g1  = (const float*)d_in[5];
    const float* be1 = (const float*)d_in[6];
    const float* m1  = (const float*)d_in[7];
    const float* v1  = (const float*)d_in[8];
    const float* w2  = (const float*)d_in[9];
    const float* b2  = (const float*)d_in[10];
    const float* g2  = (const float*)d_in[11];
    const float* be2 = (const float*)d_in[12];
    const float* m2  = (const float*)d_in[13];
    const float* v2  = (const float*)d_in[14];
    const float* w3  = (const float*)d_in[15];
    const float* b3  = (const float*)d_in[16];
    const float* g3  = (const float*)d_in[17];
    const float* be3 = (const float*)d_in[18];
    const float* m3  = (const float*)d_in[19];
    const float* v3  = (const float*)d_in[20];
    const float* wd  = (const float*)d_in[21];
    const float* bd  = (const float*)d_in[22];
    float* out = (float*)d_out;

    // f16 workspace layout (element offsets); peak ~48.5 MB, L2-resident.
    _Float16* h0  = (_Float16*)d_ws;                     // 32768 x 128
    _Float16* h1  = h0  + (size_t)32768 * 128;           // 32768 x 256
    _Float16* h2  = h1  + (size_t)32768 * 256;           // 32768 x 256
    _Float16* xT  = h2  + (size_t)32768 * 256;           // 32 x [128 x 1024]
    _Float16* Wt1 = xT  + (size_t)32 * 128 * 1024;       // 256 x 128
    _Float16* Wt2 = Wt1 + (size_t)256 * 128;             // 256 x 256
    _Float16* Wt3 = Wt2 + (size_t)256 * 256;             // 256 x 256
    _Float16* h3  = h1;                                  // reuse (h1 dead after L2)

    // Setup transposes: x [b][n][f] -> xT [b][f][n]; w [k][o] -> Wt [o][k]
    cvt_transpose_kernel<<<dim3(32, 4, 32), 256, 0, stream>>>(
        x, xT, 1024, 128, (size_t)1024 * 128, (size_t)128 * 1024);
    cvt_transpose_kernel<<<dim3(4, 8, 1), 256, 0, stream>>>(w1, Wt1, 128, 256, 0, 0);
    cvt_transpose_kernel<<<dim3(8, 8, 1), 256, 0, stream>>>(w2, Wt2, 256, 256, 0, 0);
    cvt_transpose_kernel<<<dim3(8, 8, 1), 256, 0, stream>>>(w3, Wt3, 256, 256, 0, 0);

    gin_agg_kernel<<<dim3(16, 2, 32), 256, 0, stream>>>(a, xT, x, eps, h0);
    mlp_layer_kernel<<<dim3(512, 4, 1), 256, 0, stream>>>(h0, 128, Wt1, b1, g1, be1, m1, v1, h1);
    mlp_layer_kernel<<<dim3(512, 4, 1), 256, 0, stream>>>(h1, 256, Wt2, b2, g2, be2, m2, v2, h2);
    mlp_layer_kernel<<<dim3(512, 4, 1), 256, 0, stream>>>(h2, 256, Wt3, b3, g3, be3, m3, v3, h3);
    pool_head_kernel<<<dim3(32, 1, 1), 256, 0, stream>>>(h3, wd, bd, out);
}
